// DeterministicLSTMSensorBasedForwardDynamics_75453985456460
// MI455X (gfx1250) — matrix operand order repaired and checked by reading, not compiled, not run on hardware
//
#include <hip/hip_runtime.h>
#include <hip/hip_bf16.h>

// ---------------------------------------------------------------------------
// DeterministicLSTMSensorBasedForwardDynamics — CDNA5 (gfx1250) WMMA kernel
//   B=4096, T=50, OBS=64, ACT=16 (D=80), H=256 (4H=1024), 5xMLP(256), OUT=64
// Strategy: bf16 WMMA (v_wmma_f32_16x16x32_bf16), f32 accumulate.
//   - workgroup = 64 batch rows, 512 threads (16 waves), loops T sequentially
//   - combined weight [Wi; 0pad; Wh] as K=352 x N=1024 bf16 (L2 resident)
//   - wave owns one 16-wide H-column tile; loops all 4 M tiles so each B
//     fragment is loaded once and reused 4x (no redundant L2 traffic)
//   - weight pointer laundered per timestep (defeats LICM spill-to-scratch),
//     and loads forced to address_space(1) so they emit 16-bit global_loads
//     (pure LOADcnt) instead of flat_load (LOADcnt+DScnt coupling)
//   - accumulators start from WMMA inline C=0; biases added in the epilogue
//     (kills the 128-VGPR bias-splat live range that caused residual spills)
//   - h double-buffered in LDS (bf16), c kept in VGPRs across all timesteps
// ---------------------------------------------------------------------------

typedef __attribute__((ext_vector_type(16))) __bf16 v16bf;
typedef __attribute__((ext_vector_type(8)))  __bf16 v8bf;
typedef __attribute__((ext_vector_type(8)))  float  v8f;

#define BT    64            // batch rows per workgroup
#define KTOT  352           // 80 (x) + 16 (zero pad) + 256 (h)
#define NZ    1024          // 4*H gate width
#define H     256
#define TT    50
#define XSTR  104           // LDS row stride (bf16) for x staging buffer (96 used)
#define HSTR  264           // LDS row stride (bf16) for h / mlp buffers (256 used)

// sizes (bf16 elements) inside workspace
#define WCAT_ELEMS (KTOT * NZ)          // 360448
#define MLPW_ELEMS (5 * H * H)          // 327680
#define WOUT_ELEMS (H * 64)             // 16384

__device__ __forceinline__ unsigned short f2bf(float f) {
  unsigned u = __float_as_uint(f);
  unsigned r = u + 0x7FFFu + ((u >> 16) & 1u);   // round-to-nearest-even
  return (unsigned short)(r >> 16);
}

// Opaque pointer laundering: defeats LICM so weight loads stay inside loops.
template <typename T>
__device__ __forceinline__ const T* launder_ptr(const T* p) {
  unsigned long long x = (unsigned long long)p;
  asm volatile("" : "+s"(x));
  return (const T*)x;
}

__device__ __forceinline__ v16bf joinfrag(v8bf lo, v8bf hi) {
  return __builtin_shufflevector(lo, hi, 0,1,2,3,4,5,6,7,8,9,10,11,12,13,14,15);
}

// A fragment (16x32 bf16, M x K) from an LDS row-major buffer.
// Lane l: M = l&15 ; lanes 0-15 hold K {0..7,16..23}, lanes 16-31 hold {8..15,24..31}.
__device__ __forceinline__ v16bf ldA(const unsigned short* buf, int rowStride,
                                     int rowBase, int kbase, int lane) {
  const int M  = lane & 15;
  const int c0 = (lane < 16) ? 0 : 8;
  const unsigned short* p = buf + (rowBase + M) * rowStride + kbase + c0;
  v8bf lo = *(const v8bf*)(p);
  v8bf hi = *(const v8bf*)(p + 16);
  return joinfrag(lo, hi);
}

// B fragment (32x16 bf16, K x N) from a global row-major [K][rowStride] matrix.
// Lane l: N = nb + (l&15); lanes 0-15 hold K {0..7,16..23}, lanes 16-31 hold {8..15,24..31}.
__device__ __forceinline__ v16bf ldB(const unsigned short* w, int rowStride,
                                     int kbase, int nb, int lane) {
  typedef __attribute__((address_space(1))) const __bf16* GP;
  const int c0 = (lane < 16) ? 0 : 8;
  const unsigned short* p = w + (size_t)(kbase + c0) * rowStride + nb + (lane & 15);
  v16bf r;
#pragma unroll
  for (int s = 0; s < 16; ++s)
    r[s] = *(GP)(p + (size_t)((s < 8) ? s : s + 8) * rowStride);
  return r;
}

__device__ __forceinline__ float ldg_f32(const float* p) {
  typedef __attribute__((address_space(1))) const float* GF;
  return *(GF)p;
}

__device__ __forceinline__ v8f wmma_bf16(v16bf a, v16bf b, v8f c) {
  return __builtin_amdgcn_wmma_f32_16x16x32_bf16(false, a, false, b, (short)0, c,
                                                 false, false);
}

__device__ __forceinline__ v8f zero8() {
  v8f v = {0.f, 0.f, 0.f, 0.f, 0.f, 0.f, 0.f, 0.f};
  return v;
}

__device__ __forceinline__ float sigm(float x) { return 1.0f / (1.0f + __expf(-x)); }
__device__ __forceinline__ float silu(float x) { return x * sigm(x); }

// Store a 16x16 f32 C tile as bf16 into an LDS row-major buffer.
// C/D layout: (VGPR r, lane l) -> M = r + 8*(l>=16), N = l&15.
__device__ __forceinline__ void storeTileBF16(unsigned short* buf, int rowStride,
                                              int mBase, int colBase, int lane, v8f v) {
  const int rowOff = mBase + ((lane >> 4) & 1) * 8;
  const int col    = colBase + (lane & 15);
#pragma unroll
  for (int r = 0; r < 8; ++r)
    buf[(rowOff + r) * rowStride + col] = f2bf(v[r]);
}

// ---------------------------------------------------------------------------
// Kernel 1: pack weights to bf16 in workspace.
//   ws[0 .. WCAT)            : [352][1024] = [Wi(80); zeros(16); Wh(256)]
//   ws[WCAT .. +MLPW)        : [5][256][256]
//   ws[WCAT+MLPW .. +WOUT)   : [256][64]
// ---------------------------------------------------------------------------
__global__ void pack_weights_kernel(const float* __restrict__ Wi,
                                    const float* __restrict__ Wh,
                                    const float* __restrict__ mlpW,
                                    const float* __restrict__ Wout,
                                    unsigned short* __restrict__ ws) {
  const int i = blockIdx.x * blockDim.x + threadIdx.x;
  if (i < WCAT_ELEMS) {
    const int r = i >> 10, c = i & 1023;
    float v = (r < 80) ? Wi[r * NZ + c] : (r < 96) ? 0.0f : Wh[(r - 96) * NZ + c];
    ws[i] = f2bf(v);
  } else if (i < WCAT_ELEMS + MLPW_ELEMS) {
    const int j = i - WCAT_ELEMS;
    ws[i] = f2bf(mlpW[j]);
  } else if (i < WCAT_ELEMS + MLPW_ELEMS + WOUT_ELEMS) {
    const int j = i - WCAT_ELEMS - MLPW_ELEMS;
    ws[i] = f2bf(Wout[j]);
  }
}

// ---------------------------------------------------------------------------
// Kernel 2: fused LSTM scan + MLP head. One workgroup = 64 batch rows.
// ---------------------------------------------------------------------------
__global__ __launch_bounds__(512) void lstm_mlp_kernel(
    const float* __restrict__ traj,    // [B][T][64]
    const float* __restrict__ act,     // [B][T][16]
    const float* __restrict__ bh,      // [1024]
    const float* __restrict__ mlpb,    // [5][256]
    const float* __restrict__ bout,    // [64]
    const unsigned short* __restrict__ Wcat,   // [352][1024] bf16
    const unsigned short* __restrict__ MW,     // [5][256][256] bf16
    const unsigned short* __restrict__ WO,     // [256][64] bf16
    float* __restrict__ out)           // [B][64]
{
  extern __shared__ __align__(16) unsigned short smem[];
  unsigned short* XA  = smem;               // [64][XSTR]  x_t (K rows 0..95, cols 80..95 = 0)
  unsigned short* HB0 = XA + BT * XSTR;     // [64][HSTR]  h ping
  unsigned short* HB1 = HB0 + BT * HSTR;    // [64][HSTR]  h pong

  const int tid  = threadIdx.x;
  const int lane = tid & 31;
  const int wv   = tid >> 5;     // wave 0..15 == owned H column tile (ncol)
  const int wg   = blockIdx.x;

  // init: zero pad cols of XA (80..95) and h0 buffer
  for (int i = tid; i < BT * 16; i += 512) {
    int r = i >> 4, c = i & 15;
    XA[r * XSTR + 80 + c] = 0;
  }
  for (int i = tid; i < BT * H; i += 512) {
    int r = i >> 8, c = i & 255;
    HB0[r * HSTR + c] = 0;
  }

  // per-lane gate biases for this wave's column tile (N = wv*16 + lane%16):
  // scalars only (4 VGPRs); added in the epilogue, not splatted into C.
  const int nbh = wv * 16 + (lane & 15);
  const float bI = ldg_f32(bh + 0 * H + nbh);
  const float bF = ldg_f32(bh + 1 * H + nbh);
  const float bG = ldg_f32(bh + 2 * H + nbh);
  const float bO = ldg_f32(bh + 3 * H + nbh);

  // cell state (per M tile) lives in VGPRs for the whole scan
  v8f cst[4];
#pragma unroll
  for (int m = 0; m < 4; ++m) cst[m] = zero8();

  unsigned cur = 0;
  __syncthreads();

  for (int t = 0; t < TT; ++t) {
    // ---- stage x_t (f32 -> bf16) into XA cols 0..79 ----
    for (int i = tid; i < BT * 80; i += 512) {
      const int r = i / 80, c = i - r * 80;
      const size_t b = (size_t)wg * BT + r;
      const float v = (c < 64) ? ldg_f32(traj + (b * TT + t) * 64 + c)
                               : ldg_f32(act + (b * TT + t) * 16 + (c - 64));
      XA[r * XSTR + c] = f2bf(v);
    }
    __syncthreads();

    const unsigned short* Hc = cur ? HB1 : HB0;
    unsigned short*       Hn = cur ? HB0 : HB1;

    // launder so the (t-invariant) weight loads are NOT hoisted + spilled
    const unsigned short* Wt = launder_ptr(Wcat);

    // accumulators: [m][gate], first K-step writes them with inline C = 0
    v8f zi[4], zf[4], zg[4], zo[4];
    {
      const v16bf b0 = ldB(Wt, NZ, 0, 0 * H + wv * 16, lane);
      const v16bf b1 = ldB(Wt, NZ, 0, 1 * H + wv * 16, lane);
      const v16bf b2 = ldB(Wt, NZ, 0, 2 * H + wv * 16, lane);
      const v16bf b3 = ldB(Wt, NZ, 0, 3 * H + wv * 16, lane);
#pragma unroll
      for (int m = 0; m < 4; ++m) {
        const v16bf a = ldA(XA, XSTR, m * 16, 0, lane);
        zi[m] = wmma_bf16(a, b0, zero8());
        zf[m] = wmma_bf16(a, b1, zero8());
        zg[m] = wmma_bf16(a, b2, zero8());
        zo[m] = wmma_bf16(a, b3, zero8());
      }
    }
#pragma unroll
    for (int ks = 1; ks < 11; ++ks) {
      const int kb = ks * 32;
      // 4 gate B fragments, loaded once, reused for all 4 M tiles
      const v16bf b0 = ldB(Wt, NZ, kb, 0 * H + wv * 16, lane);
      const v16bf b1 = ldB(Wt, NZ, kb, 1 * H + wv * 16, lane);
      const v16bf b2 = ldB(Wt, NZ, kb, 2 * H + wv * 16, lane);
      const v16bf b3 = ldB(Wt, NZ, kb, 3 * H + wv * 16, lane);
#pragma unroll
      for (int m = 0; m < 4; ++m) {
        const v16bf a = (ks < 3) ? ldA(XA, XSTR, m * 16, kb, lane)
                                 : ldA(Hc, HSTR, m * 16, kb - 96, lane);
        zi[m] = wmma_bf16(a, b0, zi[m]);
        zf[m] = wmma_bf16(a, b1, zf[m]);
        zg[m] = wmma_bf16(a, b2, zg[m]);
        zo[m] = wmma_bf16(a, b3, zo[m]);
      }
    }

#pragma unroll
    for (int m = 0; m < 4; ++m) {
      v8f cold = cst[m], cnew, hnew;
#pragma unroll
      for (int r = 0; r < 8; ++r) {
        const float iv = sigm(zi[m][r] + bI);
        const float fv = sigm(zf[m][r] + bF);
        const float gv = silu(zg[m][r] + bG);
        const float ov = sigm(zo[m][r] + bO);
        const float cc = fv * cold[r] + iv * gv;
        cnew[r] = cc;
        hnew[r] = ov * silu(cc);
      }
      cst[m] = cnew;
      storeTileBF16(Hn, HSTR, m * 16, wv * 16, lane, hnew);
    }
    __syncthreads();
    cur ^= 1;
  }

  // ---- MLP head: 5 x (256x256 + silu), ping-pong between HB buffers ----
  unsigned short* bufA = cur ? HB1 : HB0;   // holds final h
  unsigned short* bufB = cur ? HB0 : HB1;

  for (int layer = 0; layer < 5; ++layer) {
    const unsigned short* Wl = launder_ptr(MW + layer * H * H);
    const int nb = wv * 16;                  // wave owns one output column tile
    const float bv = ldg_f32(mlpb + layer * H + nb + (lane & 15));
    v8f acc[4];
    {
      const v16bf b = ldB(Wl, H, 0, nb, lane);
#pragma unroll
      for (int m = 0; m < 4; ++m)
        acc[m] = wmma_bf16(ldA(bufA, HSTR, m * 16, 0, lane), b, zero8());
    }
#pragma unroll
    for (int ks = 1; ks < 8; ++ks) {
      const v16bf b = ldB(Wl, H, ks * 32, nb, lane);   // loaded once, used 4x
#pragma unroll
      for (int m = 0; m < 4; ++m)
        acc[m] = wmma_bf16(ldA(bufA, HSTR, m * 16, ks * 32, lane), b, acc[m]);
    }
#pragma unroll
    for (int m = 0; m < 4; ++m) {
      v8f hv;
#pragma unroll
      for (int r = 0; r < 8; ++r) hv[r] = silu(acc[m][r] + bv);
      storeTileBF16(bufB, HSTR, m * 16, nb, lane, hv);
    }
    __syncthreads();
    unsigned short* tmp = bufA; bufA = bufB; bufB = tmp;
  }

  // ---- output layer: [64x256] @ [256x64] + bout -> d_out (f32) ----
  {
    const int m  = wv & 3;                    // 16 waves cover 4 m x 4 n tiles
    const int nb = (wv >> 2) * 16;
    const unsigned short* WOl = launder_ptr(WO);
    const float bv = ldg_f32(bout + nb + (lane & 15));
    v8f acc = wmma_bf16(ldA(bufA, HSTR, m * 16, 0, lane),
                        ldB(WOl, 64, 0, nb, lane), zero8());
#pragma unroll
    for (int ks = 1; ks < 8; ++ks)
      acc = wmma_bf16(ldA(bufA, HSTR, m * 16, ks * 32, lane),
                      ldB(WOl, 64, ks * 32, nb, lane), acc);
    const int rowOff = m * 16 + ((lane >> 4) & 1) * 8;
    const int col    = nb + (lane & 15);
#pragma unroll
    for (int r = 0; r < 8; ++r)
      out[((size_t)wg * BT + rowOff + r) * 64 + col] = acc[r] + bv;
  }
}

// ---------------------------------------------------------------------------
extern "C" void kernel_launch(void* const* d_in, const int* in_sizes, int n_in,
                              void* d_out, int out_size, void* d_ws, size_t ws_size,
                              hipStream_t stream) {
  (void)in_sizes; (void)n_in; (void)out_size; (void)ws_size;
  const float* traj = (const float*)d_in[0];
  const float* act  = (const float*)d_in[1];
  const float* Wi   = (const float*)d_in[2];
  const float* Wh   = (const float*)d_in[3];
  const float* bh   = (const float*)d_in[4];
  const float* mlpW = (const float*)d_in[5];
  const float* mlpb = (const float*)d_in[6];
  const float* Wout = (const float*)d_in[7];
  const float* bout = (const float*)d_in[8];
  unsigned short* ws = (unsigned short*)d_ws;

  const int totalConv = WCAT_ELEMS + MLPW_ELEMS + WOUT_ELEMS;
  hipLaunchKernelGGL(pack_weights_kernel, dim3((totalConv + 255) / 256), dim3(256),
                     0, stream, Wi, Wh, mlpW, Wout, ws);

  const size_t ldsBytes = (size_t)(BT * XSTR + 2 * BT * HSTR) * sizeof(unsigned short);
  hipLaunchKernelGGL(lstm_mlp_kernel, dim3(4096 / BT), dim3(512), ldsBytes, stream,
                     traj, act, bh, mlpb, bout,
                     ws,                                   // Wcat
                     ws + WCAT_ELEMS,                      // MLP weights
                     ws + WCAT_ELEMS + MLPW_ELEMS,         // Wout
                     (float*)d_out);
}
